// Attention_85005992722686
// MI455X (gfx1250) — matrix operand order repaired and checked by reading, not compiled, not run on hardware
//
#include <hip/hip_runtime.h>

#define BB 2
#define NN 2048
#define DD 1024
#define HH 16
#define DHD 64
#define MM (BB*NN)   // 4096 rows total

typedef __attribute__((ext_vector_type(16))) __bf16 v16bf;
typedef __attribute__((ext_vector_type(8)))  __bf16 v8bf;
typedef __attribute__((ext_vector_type(8)))  float  v8f;

// Load a 16-element bf16 fragment as two contiguous 16B chunks.
static __device__ __forceinline__ v16bf ld16(const __bf16* p0, const __bf16* p1) {
    v8bf lo = *(const v8bf*)p0;
    v8bf hi = *(const v8bf*)p1;
    return __builtin_shufflevector(lo, hi, 0,1,2,3,4,5,6,7,8,9,10,11,12,13,14,15);
}

static __device__ __forceinline__ v8f wmma_bf16(v16bf a, v16bf b, v8f c) {
    // D = A(16x32) * B(32x16) + C, fp32 accumulate
    return __builtin_amdgcn_wmma_f32_16x16x32_bf16(false, a, false, b, (short)0, c, false, false);
}

static __device__ __forceinline__ float redmax16(float x) {
    x = fmaxf(x, __shfl_xor(x, 1, 16));
    x = fmaxf(x, __shfl_xor(x, 2, 16));
    x = fmaxf(x, __shfl_xor(x, 4, 16));
    x = fmaxf(x, __shfl_xor(x, 8, 16));
    return x;
}
static __device__ __forceinline__ float redsum16(float x) {
    x += __shfl_xor(x, 1, 16);
    x += __shfl_xor(x, 2, 16);
    x += __shfl_xor(x, 4, 16);
    x += __shfl_xor(x, 8, 16);
    return x;
}

// ---------------- LayerNorm: fp32 row -> two bf16 rows -----------------
__global__ __launch_bounds__(256)
void ln_kernel(const float* __restrict__ x,
               const float* __restrict__ g1, const float* __restrict__ b1,
               const float* __restrict__ g2, const float* __restrict__ b2,
               __bf16* __restrict__ xn, __bf16* __restrict__ cn) {
    const int tid = threadIdx.x;
    const size_t row = blockIdx.x;
    const float4 v = ((const float4*)(x + row * DD))[tid];
    float s = v.x + v.y + v.z + v.w;
    float q = v.x*v.x + v.y*v.y + v.z*v.z + v.w*v.w;
#pragma unroll
    for (int m = 1; m < 32; m <<= 1) { s += __shfl_xor(s, m, 32); q += __shfl_xor(q, m, 32); }
    __shared__ float ssum[8], ssq[8];
    if ((tid & 31) == 0) { ssum[tid >> 5] = s; ssq[tid >> 5] = q; }
    __syncthreads();
    float S = 0.f, Q = 0.f;
#pragma unroll
    for (int i = 0; i < 8; ++i) { S += ssum[i]; Q += ssq[i]; }
    const float mean = S * (1.0f / DD);
    const float var  = Q * (1.0f / DD) - mean * mean;
    const float rstd = rsqrtf(var + 1e-5f);
    const float4 ga = ((const float4*)g1)[tid], ba = ((const float4*)b1)[tid];
    const float4 gc = ((const float4*)g2)[tid], bc = ((const float4*)b2)[tid];
    const float xv[4] = {v.x, v.y, v.z, v.w};
    const float gav[4] = {ga.x, ga.y, ga.z, ga.w}, bav[4] = {ba.x, ba.y, ba.z, ba.w};
    const float gcv[4] = {gc.x, gc.y, gc.z, gc.w}, bcv[4] = {bc.x, bc.y, bc.z, bc.w};
#pragma unroll
    for (int j = 0; j < 4; ++j) {
        const float nv = (xv[j] - mean) * rstd;
        xn[row * DD + 4 * tid + j] = (__bf16)(nv * gav[j] + bav[j]);
        cn[row * DD + 4 * tid + j] = (__bf16)(nv * gcv[j] + bcv[j]);
    }
}

// ------------- fp32 [K][N] -> bf16 transposed [N][K] -------------------
__global__ __launch_bounds__(256)
void cvtT_kernel(const float* __restrict__ W, __bf16* __restrict__ WT, int K, int Ncol) {
    const size_t i = (size_t)blockIdx.x * blockDim.x + threadIdx.x;
    if (i < (size_t)K * Ncol) {
        const int k = (int)(i / Ncol), n = (int)(i % Ncol);
        WT[(size_t)n * K + k] = (__bf16)W[i];
    }
}

// ------------- one wave computes a 16x64 tile of A @ B -----------------
// A: row-major [M][K] bf16.  BT: column-major weights, i.e. [Ncol][K] bf16.
static __device__ __forceinline__ void wave_gemm_16x64(
        const __bf16* __restrict__ A, const __bf16* __restrict__ BT,
        int K, int row0, int col0, v8f acc[4]) {
    const int lane = threadIdx.x & 31, half = lane >> 4, lq = lane & 15;
    const __bf16* arow = A + (size_t)(row0 + lq) * K;
    const __bf16* bp0 = BT + (size_t)(col0 +  0 + lq) * K + 16 * half;
    const __bf16* bp1 = BT + (size_t)(col0 + 16 + lq) * K + 16 * half;
    const __bf16* bp2 = BT + (size_t)(col0 + 32 + lq) * K + 16 * half;
    const __bf16* bp3 = BT + (size_t)(col0 + 48 + lq) * K + 16 * half;
    v8f z = {};
    acc[0] = z; acc[1] = z; acc[2] = z; acc[3] = z;
    for (int kk = 0; kk < K; kk += 32) {
        __builtin_prefetch(arow + kk + 64, 0, 3);
        const v16bf a = ld16(arow + kk + 16 * half, arow + kk + 16 * half + 8);
        acc[0] = wmma_bf16(a, ld16(bp0 + kk, bp0 + kk + 8), acc[0]);
        acc[1] = wmma_bf16(a, ld16(bp1 + kk, bp1 + kk + 8), acc[1]);
        acc[2] = wmma_bf16(a, ld16(bp2 + kk, bp2 + kk + 8), acc[2]);
        acc[3] = wmma_bf16(a, ld16(bp3 + kk, bp3 + kk + 8), acc[3]);
    }
}

// ---------------- Q projection: xn @ Wq -> head-major bf16 -------------
__global__ __launch_bounds__(256)
void proj_q_kernel(const __bf16* __restrict__ xn, const __bf16* __restrict__ WqT,
                   __bf16* __restrict__ Qh) {
    const int wid = (int)((blockIdx.x * blockDim.x + threadIdx.x) >> 5);
    const int rt = wid >> 4, ct = wid & 15;           // 256 row tiles x 16 col tiles
    const int row0 = rt * 16, col0 = ct * 64;
    v8f acc[4];
    wave_gemm_16x64(xn, WqT, DD, row0, col0, acc);
    const int lane = threadIdx.x & 31, half = lane >> 4, lq = lane & 15;
    const int bbi = row0 / NN, tok0 = row0 % NN;
    const size_t bh = (size_t)bbi * HH + ct;          // head == ct
#pragma unroll
    for (int t = 0; t < 4; ++t)
#pragma unroll
        for (int r = 0; r < 8; ++r) {
            const int tok = tok0 + r + 8 * half;
            Qh[(bh * NN + tok) * DHD + t * 16 + lq] = (__bf16)(acc[t][r] * 0.125f); // DH^-0.5
        }
}

// ------------- KV projection: cn @ Wkv -> K head-major, V dim-major ----
__global__ __launch_bounds__(256)
void proj_kv_kernel(const __bf16* __restrict__ cn, const __bf16* __restrict__ WkvT,
                    __bf16* __restrict__ Kh, __bf16* __restrict__ Vt) {
    const int wid = (int)((blockIdx.x * blockDim.x + threadIdx.x) >> 5);
    const int rt = wid >> 5, ct = wid & 31;           // 256 row tiles x 32 col tiles
    const int row0 = rt * 16, col0 = ct * 64;
    v8f acc[4];
    wave_gemm_16x64(cn, WkvT, DD, row0, col0, acc);
    const int lane = threadIdx.x & 31, half = lane >> 4, lq = lane & 15;
    const int bbi = row0 / NN, tok0 = row0 % NN;
    if (ct < 16) {
        const size_t bh = (size_t)bbi * HH + ct;
#pragma unroll
        for (int t = 0; t < 4; ++t)
#pragma unroll
            for (int r = 0; r < 8; ++r) {
                const int tok = tok0 + r + 8 * half;
                Kh[(bh * NN + tok) * DHD + t * 16 + lq] = (__bf16)acc[t][r];
            }
    } else {
        const size_t bh = (size_t)bbi * HH + (ct - 16);
#pragma unroll
        for (int t = 0; t < 4; ++t)
#pragma unroll
            for (int r = 0; r < 8; ++r) {
                const int tok = tok0 + r + 8 * half;
                Vt[(bh * DHD + t * 16 + lq) * NN + tok] = (__bf16)acc[t][r];  // dim-major
            }
    }
}

// -------- flash attention: one 32-thread block (1 wave) per Q tile -----
__global__ __launch_bounds__(32)
void attn_kernel(const __bf16* __restrict__ Qh, const __bf16* __restrict__ Kh,
                 const __bf16* __restrict__ Vt, __bf16* __restrict__ Ob) {
    __shared__ __bf16 Pls[16][32];                  // P tile, C-layout -> A-layout via LDS
    const int bid = blockIdx.x;
    const int qt = bid & 127;                        // N/16 = 128 Q tiles
    const int h  = (bid >> 7) & 15;
    const int bb = bid >> 11;
    const int lane = threadIdx.x & 31, half = lane >> 4, lq = lane & 15;
    const size_t bh = (size_t)bb * HH + h;

    const __bf16* qrow = Qh + (bh * NN + qt * 16 + lq) * DHD;
    const v16bf aQ0 = ld16(qrow + 16 * half,      qrow + 8 + 16 * half);  // k 0..31
    const v16bf aQ1 = ld16(qrow + 32 + 16 * half, qrow + 40 + 16 * half);  // k 32..63

    float mrow[8], lrow[8];
    v8f o[4]; v8f z = {};
#pragma unroll
    for (int r = 0; r < 8; ++r) { mrow[r] = -3.0e38f; lrow[r] = 0.f; }
#pragma unroll
    for (int t = 0; t < 4; ++t) o[t] = z;

    const int rowbase = qt * 16 + 8 * half;
    for (int j0 = 0; j0 < (qt + 1) * 16; j0 += 32) {
        // S = Q @ K^T for 32 key tokens (two 16-col C fragments)
        const __bf16* k0 = Kh + (bh * NN + j0 + lq) * DHD + 16 * half;
        const __bf16* k1 = Kh + (bh * NN + j0 + 16 + lq) * DHD + 16 * half;
        v8f s0 = z, s1 = z;
        s0 = wmma_bf16(aQ0, ld16(k0, k0 + 8),       s0);
        s0 = wmma_bf16(aQ1, ld16(k0 + 32, k0 + 40), s0);
        s1 = wmma_bf16(aQ0, ld16(k1, k1 + 8),       s1);
        s1 = wmma_bf16(aQ1, ld16(k1 + 32, k1 + 40), s1);

        float alpha[8];
#pragma unroll
        for (int r = 0; r < 8; ++r) {
            const int row = rowbase + r;
            float sv0 = s0[r], sv1 = s1[r];
            if (j0 + lq > row)      sv0 = -1.0e30f;   // causal mask
            if (j0 + 16 + lq > row) sv1 = -1.0e30f;
            const float mx = redmax16(fmaxf(sv0, sv1));
            const float mnew = fmaxf(mrow[r], mx);
            alpha[r] = __expf(mrow[r] - mnew);
            const float p0 = __expf(sv0 - mnew);
            const float p1 = __expf(sv1 - mnew);
            lrow[r] = lrow[r] * alpha[r] + redsum16(p0 + p1);
            mrow[r] = mnew;
            Pls[r + 8 * half][lq]      = (__bf16)p0;
            Pls[r + 8 * half][16 + lq] = (__bf16)p1;
        }
#pragma unroll
        for (int t = 0; t < 4; ++t)
#pragma unroll
            for (int r = 0; r < 8; ++r) o[t][r] *= alpha[r];

        __syncthreads();  // single-wave block: cheap; orders LDS P write -> read
        const v16bf aP = ld16(&Pls[lq][16 * half], &Pls[lq][16 * half + 8]);

        // O += P @ V  (V dim-major -> contiguous B fragments)
#pragma unroll
        for (int t = 0; t < 4; ++t) {
            const __bf16* vp = Vt + (bh * DHD + t * 16 + lq) * NN + j0 + 16 * half;
            o[t] = wmma_bf16(aP, ld16(vp, vp + 8), o[t]);
        }
        __syncthreads();
    }

#pragma unroll
    for (int t = 0; t < 4; ++t)
#pragma unroll
        for (int r = 0; r < 8; ++r) {
            const float val = o[t][r] / lrow[r];
            Ob[((size_t)bb * NN + qt * 16 + r + 8 * half) * DD + h * DHD + t * 16 + lq] = (__bf16)val;
        }
}

// ---------------- output projection: Ob @ Wo -> fp32 out ---------------
__global__ __launch_bounds__(256)
void proj_o_kernel(const __bf16* __restrict__ Ob, const __bf16* __restrict__ WoT,
                   float* __restrict__ out) {
    const int wid = (int)((blockIdx.x * blockDim.x + threadIdx.x) >> 5);
    const int rt = wid >> 4, ct = wid & 15;
    const int row0 = rt * 16, col0 = ct * 64;
    v8f acc[4];
    wave_gemm_16x64(Ob, WoT, DD, row0, col0, acc);
    const int lane = threadIdx.x & 31, half = lane >> 4, lq = lane & 15;
#pragma unroll
    for (int t = 0; t < 4; ++t)
#pragma unroll
        for (int r = 0; r < 8; ++r)
            out[(size_t)(row0 + r + 8 * half) * DD + col0 + t * 16 + lq] = acc[t][r];
}

extern "C" void kernel_launch(void* const* d_in, const int* in_sizes, int n_in,
                              void* d_out, int out_size, void* d_ws, size_t ws_size,
                              hipStream_t stream) {
    const float* x     = (const float*)d_in[0];
    const float* ln_g  = (const float*)d_in[1];
    const float* ln_b  = (const float*)d_in[2];
    const float* lnc_g = (const float*)d_in[3];
    const float* lnc_b = (const float*)d_in[4];
    const float* Wq    = (const float*)d_in[5];
    const float* Wkv   = (const float*)d_in[6];
    const float* Wo    = (const float*)d_in[7];
    float* out = (float*)d_out;

    // workspace carve-up (~59 MB total)
    char* ws = (char*)d_ws; size_t off = 0;
    auto take = [&](size_t bytes) -> char* {
        char* p = ws + off; off += (bytes + 255) & ~(size_t)255; return p;
    };
    __bf16* xn   = (__bf16*)take((size_t)MM * DD * 2);
    __bf16* cn   = (__bf16*)take((size_t)MM * DD * 2);
    __bf16* WqT  = (__bf16*)take((size_t)DD * DD * 2);
    __bf16* WkvT = (__bf16*)take((size_t)DD * 2 * DD * 2);
    __bf16* WoT  = (__bf16*)take((size_t)DD * DD * 2);
    __bf16* Qh   = (__bf16*)take((size_t)MM * DD * 2);   // [B,H,N,64]
    __bf16* Kh   = (__bf16*)take((size_t)MM * DD * 2);   // [B,H,N,64]
    __bf16* Vt   = (__bf16*)take((size_t)MM * DD * 2);   // [B,H,64,N]
    __bf16* Ob   = (__bf16*)take((size_t)MM * DD * 2);   // [B*N, H*64]

    ln_kernel<<<MM, 256, 0, stream>>>(x, ln_g, ln_b, lnc_g, lnc_b, xn, cn);
    cvtT_kernel<<<(DD * DD) / 256, 256, 0, stream>>>(Wq, WqT, DD, DD);
    cvtT_kernel<<<(DD * 2 * DD) / 256, 256, 0, stream>>>(Wkv, WkvT, DD, 2 * DD);
    cvtT_kernel<<<(DD * DD) / 256, 256, 0, stream>>>(Wo, WoT, DD, DD);
    proj_q_kernel<<<(4096 * 32) / 256, 256, 0, stream>>>(xn, WqT, Qh);
    proj_kv_kernel<<<(8192 * 32) / 256, 256, 0, stream>>>(cn, WkvT, Kh, Vt);
    attn_kernel<<<BB * HH * (NN / 16), 32, 0, stream>>>(Qh, Kh, Vt, Ob);
    proj_o_kernel<<<(4096 * 32) / 256, 256, 0, stream>>>(Ob, WoT, out);
}
